// AdditiveAttention_38826504356464
// MI455X (gfx1250) — matrix operand order repaired and checked by reading, not compiled, not run on hardware
//
#include <hip/hip_runtime.h>
#include <cmath>

// Problem constants (from reference)
#define B_ 4
#define Q_ 256
#define K_ 256
#define H_ 512
#define NEG_ (-1e9f)

// GEMM macro-tile per wave: 32 (M) x 32 (N) -> 2x2 WMMA tiles, 4 WMMAs/K-step.
#define TM_ 32
#define TN_ 32

typedef __attribute__((ext_vector_type(16))) __bf16 v16bf;
typedef __attribute__((ext_vector_type(8)))  float  v8f;

// Hardware tanh if this toolchain declares the builtin (CDNA5 V_TANH_F32,
// a TRANS op that co-executes with VALU); OCML polynomial otherwise.
static __device__ __forceinline__ float fast_tanh(float x) {
#if __has_builtin(__builtin_amdgcn_tanhf)
    return __builtin_amdgcn_tanhf(x);
#elif __has_builtin(__builtin_amdgcn_tanh_f32)
    return __builtin_amdgcn_tanh_f32(x);
#else
    return tanhf(x);
#endif
}

// ---------------------------------------------------------------------------
// Batched GEMM: C[z] = A[z] (MxKD, row-major f32) * B[z] (KDxNN, row-major f32)
// One wave32 per 32x32 output macro-tile: 2 A-fragments x 2 B-fragments ->
// 4 V_WMMA_F32_16X16X32_BF16 per K-step (f32 accumulate), independent
// accumulators. f32->bf16 via native v_cvt_pk_bf16_f32 ((__bf16) casts).
// A-fragment rows load as float4 (global_load_b128); B column strides and C
// store strides are compile-time (template NN) so they fold into immediate
// IOFFSETs. __launch_bounds__(32,1) grants the full VGPR budget (no spills).
//
// Fragment packing used here (contiguous k order, same for A and B):
//   A (16-bit 16x32): lane<16 holds K {0..15}, lane>=16 holds {16..31}
//   B (32x16 KxN):    per-lane column l16; half selects K base 0 / 16
//   C/D: VGPR r -> M = half*8 + r, N = l16
// EXEC is all-ones (blockDim=32, exact tiling, no divergence) as WMMA requires.
// ---------------------------------------------------------------------------
template <int KD, int NN>
__global__ __launch_bounds__(32, 1) void wmma_gemm_bf16(
        const float* __restrict__ A,
        const float* __restrict__ Bm,
        float* __restrict__ C,
        long sA, long sB, long sC) {
    const float* Ab = A  + (size_t)blockIdx.z * (size_t)sA;
    const float* Bb = Bm + (size_t)blockIdx.z * (size_t)sB;
    float*       Cb = C  + (size_t)blockIdx.z * (size_t)sC;

    const int tn   = blockIdx.x * TN_;    // output column macro-tile
    const int tm   = blockIdx.y * TM_;    // output row macro-tile
    const int lane = threadIdx.x;         // 0..31 (wave32)
    const int half = lane >> 4;           // 0 | 1
    const int l16  = lane & 15;

    v8f acc[2][2] = {};

    for (int kk = 0; kk < KD; kk += 32) {
        // A fragments for the two 16-row sub-tiles (vectorized b128 loads).
        v16bf af[2];
#pragma unroll
        for (int mi = 0; mi < 2; ++mi) {
            const float* arow =
                Ab + (size_t)(tm + mi * 16 + l16) * KD + kk + half * 16;
            const float4 a0 = *(const float4*)(arow);        // K +0..3
            const float4 a1 = *(const float4*)(arow + 4);    // K +4..7
            const float4 a2 = *(const float4*)(arow + 8);    // K +8..11
            const float4 a3 = *(const float4*)(arow + 12);   // K +12..15
            af[mi][0]  = (__bf16)a0.x; af[mi][1]  = (__bf16)a0.y;
            af[mi][2]  = (__bf16)a0.z; af[mi][3]  = (__bf16)a0.w;
            af[mi][4]  = (__bf16)a1.x; af[mi][5]  = (__bf16)a1.y;
            af[mi][6]  = (__bf16)a1.z; af[mi][7]  = (__bf16)a1.w;
            af[mi][8]  = (__bf16)a2.x; af[mi][9]  = (__bf16)a2.y;
            af[mi][10] = (__bf16)a2.z; af[mi][11] = (__bf16)a2.w;
            af[mi][12] = (__bf16)a3.x; af[mi][13] = (__bf16)a3.y;
            af[mi][14] = (__bf16)a3.z; af[mi][15] = (__bf16)a3.w;
        }

        // B fragments for the two 16-column sub-tiles: single base pointer,
        // compile-time stride NN -> immediate offsets, coalesced across lanes,
        // L2-resident.
        v16bf bf[2];
#pragma unroll
        for (int ni = 0; ni < 2; ++ni) {
            const float* bcol =
                Bb + (size_t)(kk + half * 16) * NN + tn + ni * 16 + l16;
#pragma unroll
            for (int j = 0; j < 16; ++j)
                bf[ni][j] = (__bf16)bcol[(size_t)j * NN];
        }

        // 4 WMMAs reusing the loaded fragments (independent accumulators).
#pragma unroll
        for (int mi = 0; mi < 2; ++mi)
#pragma unroll
            for (int ni = 0; ni < 2; ++ni)
                acc[mi][ni] = __builtin_amdgcn_wmma_f32_16x16x32_bf16(
                    /*neg_a=*/false, af[mi], /*neg_b=*/false, bf[ni],
                    /*c_mod=*/(short)0, acc[mi][ni],
                    /*reuse_a=*/false, /*reuse_b=*/false);
    }

#pragma unroll
    for (int mi = 0; mi < 2; ++mi) {
#pragma unroll
        for (int ni = 0; ni < 2; ++ni) {
            float* yout = Cb + (size_t)(tm + mi * 16 + half * 8) * NN
                             + tn + ni * 16 + l16;
#pragma unroll
            for (int r = 0; r < 8; ++r)
                yout[(size_t)r * NN] = acc[mi][ni][r];   // imm offsets (r*NN*4)
        }
    }
}

// ---------------------------------------------------------------------------
// Fused additive score + masked softmax.
// Grid: (Q_, B_), block: K_ = 256 threads; thread t owns key index k = t.
// scores[b,q,k] = sum_h wv[h] * tanh(qp[b,q,h] + kp[b,k,h])   (TRANS-bound)
// kp is tiled through LDS in 256x32 chunks with 33-float padded rows (33
// coprime with the 64 LDS banks -> conflict-free [t][j] reads). Each thread
// prefetches the cacheline of its row's *next* chunk (global_prefetch_b8)
// while computing the current one.
// ---------------------------------------------------------------------------
__global__ void additive_scores_softmax(const float* __restrict__ qp,
                                        const float* __restrict__ kp,
                                        const float* __restrict__ wv,
                                        const int*   __restrict__ vlen,
                                        float* __restrict__ attn) {
    __shared__ float s_q[H_];          // 2 KB
    __shared__ float s_wv[H_];         // 2 KB
    __shared__ float s_kp[K_ * 33];    // 33 KB (padded rows)
    __shared__ float red[K_];          // 1 KB

    const int b = blockIdx.y;
    const int q = blockIdx.x;
    const int t = threadIdx.x;         // key index k

    for (int h = t; h < H_; h += K_) {
        s_q[h]  = qp[((size_t)(b * Q_ + q)) * H_ + h];
        s_wv[h] = wv[h];
    }
    __syncthreads();

    float acc = 0.0f;
    for (int h0 = 0; h0 < H_; h0 += 32) {
        // Cooperative, coalesced load of kp[b, 0..255, h0..h0+31] into LDS.
        for (int idx = t; idx < K_ * 32; idx += K_) {
            const int kk = idx >> 5;
            const int j  = idx & 31;
            s_kp[kk * 33 + j] = kp[((size_t)(b * K_ + kk)) * H_ + h0 + j];
        }
        __syncthreads();

        // Prefetch next chunk: thread t pulls the cacheline holding
        // kp[b, t, h0+32 .. h0+63] (128 B = exactly one chunk row).
        if (h0 + 32 < H_)
            __builtin_prefetch(&kp[((size_t)(b * K_ + t)) * H_ + h0 + 32], 0, 3);

#pragma unroll
        for (int j = 0; j < 32; ++j)
            acc = __builtin_fmaf(s_wv[h0 + j],
                                 fast_tanh(s_q[h0 + j] + s_kp[t * 33 + j]),
                                 acc);
        __syncthreads();
    }

    // Sequence mask: k >= valid_len -> -1e9 (valid_len==0 -> uniform softmax,
    // matching jax.nn.softmax over an all-masked row).
    const int vl = vlen[b];
    const float score = (t < vl) ? acc : NEG_;

    // Block softmax over K_=256 values.
    red[t] = score;
    __syncthreads();
    for (int s = K_ / 2; s > 0; s >>= 1) {
        if (t < s) red[t] = fmaxf(red[t], red[t + s]);
        __syncthreads();
    }
    const float mx = red[0];
    __syncthreads();

    const float e = __expf(score - mx);
    red[t] = e;
    __syncthreads();
    for (int s = K_ / 2; s > 0; s >>= 1) {
        if (t < s) red[t] += red[t + s];
        __syncthreads();
    }
    const float inv = 1.0f / red[0];

    attn[((size_t)(b * Q_ + q)) * K_ + t] = e * inv;
}

extern "C" void kernel_launch(void* const* d_in, const int* in_sizes, int n_in,
                              void* d_out, int out_size, void* d_ws, size_t ws_size,
                              hipStream_t stream) {
    (void)in_sizes; (void)n_in; (void)out_size; (void)ws_size;

    const float* query = (const float*)d_in[0];   // [B,Q,H]
    const float* key   = (const float*)d_in[1];   // [B,K,H]
    const float* value = (const float*)d_in[2];   // [B,K,H]
    const int*   vlens = (const int*)  d_in[3];   // [B]
    const float* Wq    = (const float*)d_in[4];   // [H,H]
    const float* Wk    = (const float*)d_in[5];   // [H,H]
    const float* wv    = (const float*)d_in[6];   // [H]
    float*       out   = (float*)d_out;           // [B,Q,H]

    // Workspace: qp | kp | attn  (f32)  -- 5 MB total
    float* ws   = (float*)d_ws;
    float* qp   = ws;                                   // B*Q*H = 524288
    float* kp   = qp + (size_t)B_ * Q_ * H_;            // B*K*H = 524288
    float* attn = kp + (size_t)B_ * K_ * H_;            // B*Q*K = 262144

    const dim3 wblk(32);

    // 1) q projection: [B*Q, H] = query @ Wq   (1024x512x512, WMMA bf16)
    wmma_gemm_bf16<H_, H_><<<dim3(H_ / TN_, (B_ * Q_) / TM_, 1), wblk, 0, stream>>>(
        query, Wq, qp, 0, 0, 0);

    // 2) k projection: [B*K, H] = key @ Wk
    wmma_gemm_bf16<H_, H_><<<dim3(H_ / TN_, (B_ * K_) / TM_, 1), wblk, 0, stream>>>(
        key, Wk, kp, 0, 0, 0);

    // 3) Fused tanh-score + mask + softmax -> attn [B,Q,K]
    additive_scores_softmax<<<dim3(Q_, B_), dim3(K_), 0, stream>>>(
        qp, kp, wv, vlens, attn);

    // 4) context[b] = attn[b] (QxK) @ value[b] (KxH)  (batched WMMA bf16)
    wmma_gemm_bf16<K_, H_><<<dim3(H_ / TN_, Q_ / TM_, B_), wblk, 0, stream>>>(
        attn, value, out,
        (long)Q_ * K_, (long)K_ * H_, (long)Q_ * H_);
}
